// PropLayer_24584392802715
// MI455X (gfx1250) — compile-verified, matrix operand order checked by reading, not run on hardware
//
#include <hip/hip_runtime.h>

// ---------- types ----------
typedef __attribute__((ext_vector_type(16))) __bf16        v16bf;
typedef __attribute__((ext_vector_type(2)))  __bf16        v2bf;
typedef __attribute__((ext_vector_type(8)))  float         v8f;
typedef __attribute__((ext_vector_type(4)))  unsigned int  u32x4;   // 16B, forces b128 ld/st
typedef __attribute__((ext_vector_type(4)))  float         f32x4;
typedef __attribute__((ext_vector_type(2)))  unsigned int  u32x2;

union Frag {
  v16bf v;
  u32x4 q[2];
};

__device__ __forceinline__ unsigned short f2bf(float f) {
  unsigned int u = __float_as_uint(f);
  u += 0x7FFFu + ((u >> 16) & 1u);   // round-to-nearest-even
  return (unsigned short)(u >> 16);
}

// packed 2xf32 -> 2xbf16 in one VALU op where available
__device__ __forceinline__ unsigned int pkbf(float lo, float hi) {
#if __has_builtin(__builtin_amdgcn_cvt_pk_bf16_f32)
  union { v2bf v; unsigned int u; } cv;
  cv.v = __builtin_amdgcn_cvt_pk_bf16_f32(lo, hi);
  return cv.u;
#else
  return (unsigned)f2bf(lo) | ((unsigned)f2bf(hi) << 16);
#endif
}

__device__ __forceinline__ void wait_async0() {
#if __has_builtin(__builtin_amdgcn_s_wait_asynccnt)
  __builtin_amdgcn_s_wait_asynccnt(0);
#else
  asm volatile("s_wait_asynccnt 0" ::: "memory");
#endif
}

#define BT   128      // batch rows per workgroup
#define PP   19       // points
#define DIM  256      // D == O == 256
#define LK   264      // LDS row stride (bf16 elems); +16B pad kills 512B-stride bank conflicts

// LDS layout (bytes)
#define XS_OFF   0                         // [128][264] bf16 aggregated input
#define HS_OFF   67584                     // [128][264] bf16 hidden
#define WS_OFF   135168                    // [256][264] bf16 weights [n][k]
#define AFF_OFF  (WS_OFF + 135168)         // [128][19] f32
#define B1_OFF   (AFF_OFF + 9728)          // [256] f32
#define B2_OFF   (B1_OFF + 1024)           // [256] f32
#define SMEM_BYTES (B2_OFF + 1024)         // 282112 <= 320K

// ---------- weight prep: f32 [j][k][n] -> bf16 [j][n][k] ----------
__global__ __launch_bounds__(256)
void prep_weights(const float* __restrict__ w, unsigned short* __restrict__ wt) {
  __shared__ float tile[32][33];
  const int j  = blockIdx.z;
  const int n0 = blockIdx.x * 32;
  const int k0 = blockIdx.y * 32;
  const int tx = threadIdx.x & 31;
  const int ty = threadIdx.x >> 5;          // 0..7
  const float* src = w + (size_t)j * DIM * DIM;
  unsigned short* dst = wt + (size_t)j * DIM * DIM;
#pragma unroll
  for (int i = 0; i < 4; ++i)
    tile[ty + 8 * i][tx] = src[(size_t)(k0 + ty + 8 * i) * DIM + n0 + tx];
  __syncthreads();
#pragma unroll
  for (int i = 0; i < 4; ++i)
    dst[(size_t)(n0 + ty + 8 * i) * DIM + k0 + tx] = f2bf(tile[tx][ty + 8 * i]);
}

// ---------- async DMA of a full W matrix (bf16 [n][k], 128KB) into LDS [n][LK] ----------
// 256 threads x 32 iters x 16B; ASYNCcnt-tracked, streams while the wave computes.
__device__ __forceinline__ void stage_w_async_full(const unsigned short* wt_j,
                                                   unsigned ldsbase, int tid) {
  const int cb = tid & 31;             // 16B column block (32 x 16B = 512B row)
  const int rw = tid >> 5;             // row 0..7 per pass
  unsigned lds  = ldsbase + (unsigned)(rw * (LK * 2) + cb * 16);
  unsigned voff = (unsigned)(rw * (DIM * 2) + cb * 16);
  const unsigned long long sbase = (unsigned long long)wt_j;
#pragma unroll
  for (int c = 0; c < 32; ++c) {
    asm volatile("global_load_async_to_lds_b128 %0, %1, %2"
                 :: "v"(lds), "v"(voff), "s"(sbase) : "memory");
    lds  += 8u * (LK * 2);
    voff += 8u * (DIM * 2);
  }
}

// ---------- GEMM helper ----------
// C(128x256,f32) += A(128x256,bf16 LDS) * W(256x256,bf16 LDS, [n][k] layout)
// Single base VGPR per operand; all mi/ni/kt offsets fold into ds imm offsets.
__device__ __forceinline__ void gemm_128x256(const unsigned short* __restrict__ Abase,
                                             const unsigned short* __restrict__ Wbase,
                                             v8f acc[16]) {
#pragma unroll
  for (int kt = 0; kt < 8; ++kt) {
    Frag bfr[4];
#pragma unroll
    for (int ni = 0; ni < 4; ++ni) {
      bfr[ni].q[0] = *(const u32x4*)(Wbase + ni * (16 * LK) + kt * 32);
      bfr[ni].q[1] = *(const u32x4*)(Wbase + ni * (16 * LK) + kt * 32 + 16);
    }
#pragma unroll
    for (int mi = 0; mi < 4; ++mi) {
      Frag af;
      af.q[0] = *(const u32x4*)(Abase + mi * (16 * LK) + kt * 32);
      af.q[1] = *(const u32x4*)(Abase + mi * (16 * LK) + kt * 32 + 16);
#pragma unroll
      for (int ni = 0; ni < 4; ++ni)
        acc[mi * 4 + ni] = __builtin_amdgcn_wmma_f32_16x16x32_bf16(
            false, af.v, false, bfr[ni].v, (short)0, acc[mi * 4 + ni], false, false);
    }
  }
}

// ---------- fused kernel ----------
__global__ __launch_bounds__(256)
void prop_fused(const float* __restrict__ inp,
                const float* __restrict__ aff,
                const float* __restrict__ bias1,
                const float* __restrict__ bias2,
                const unsigned short* __restrict__ wt1,
                const unsigned short* __restrict__ wt2,
                float* __restrict__ out) {
  extern __shared__ char smem[];
  unsigned short* Xs = (unsigned short*)(smem + XS_OFF);
  unsigned short* Hs = (unsigned short*)(smem + HS_OFF);
  unsigned short* Ws = (unsigned short*)(smem + WS_OFF);
  float* affs = (float*)(smem + AFF_OFF);
  float* b1s  = (float*)(smem + B1_OFF);
  float* b2s  = (float*)(smem + B2_OFF);
  const unsigned smem0 = (unsigned)(unsigned long long)(void*)smem;  // LDS byte base

  const int tid  = threadIdx.x;
  const int lane = tid & 31;
  const int wave = tid >> 5;           // 0..7
  const int ln   = lane & 15;
  const int hi   = lane >> 4;          // K-half selector for fragments
  const int j    = blockIdx.x;         // point index fastest -> 19 WGs share one inp tile in L2
  const long b0  = (long)blockIdx.y * BT;

  // ---- kick off W1 DMA: 128KB streams into LDS during the whole aggregation phase ----
  stage_w_async_full(wt1 + (size_t)j * DIM * DIM, smem0 + WS_OFF, tid);

  // ---- stage biases + affinity tile ----
  b1s[tid] = bias1[j * DIM + tid];
  b2s[tid] = bias2[j * DIM + tid];
  for (int idx = tid; idx < BT * PP; idx += 256) {
    int r = idx / PP, k = idx - r * PP;
    affs[idx] = aff[(b0 + r) * (PP * PP) + j * PP + k];
  }
  __syncthreads();                     // affs/bias visible (W1 still streaming)

  // ---- phase 1: affinity aggregation (VALU f32) + inp passthrough ----
  const int rbase = wave * 16;
  const int c0 = lane * 4;
  f32x4 acc0[16], acc1[16];
  {
    const f32x4 z = {0.f, 0.f, 0.f, 0.f};
#pragma unroll
    for (int r = 0; r < 16; ++r) { acc0[r] = z; acc1[r] = z; }
  }
  const float* affbase = affs + rbase * PP;

  for (int k = 0; k < PP; ++k) {
    const float* src0 = inp + ((b0 + rbase) * PP + k) * DIM + c0;
    const bool pass = (k == j);
#pragma unroll
    for (int r = 0; r < 16; ++r) {
      f32x4 v0 = *(const f32x4*)(src0 + r * (PP * DIM));
      f32x4 v1 = *(const f32x4*)(src0 + r * (PP * DIM) + 128);
      const float s = affbase[r * PP + k];
      acc0[r] += s * v0;
      acc1[r] += s * v1;
      if (pass) {                      // out[..., 0:256] = inp (uniform branch)
        float* d0 = out + ((b0 + rbase) * PP + j) * 512 + c0;
        *(f32x4*)(d0 + r * (PP * 512)) = v0;
        *(f32x4*)(d0 + r * (PP * 512) + 128) = v1;
      }
    }
  }
  // store X to LDS as bf16
  {
    unsigned short* xb = Xs + rbase * LK + c0;
#pragma unroll
    for (int r = 0; r < 16; ++r) {
      u32x2 p;
      p.x = pkbf(acc0[r][0], acc0[r][1]);
      p.y = pkbf(acc0[r][2], acc0[r][3]);
      *(u32x2*)(xb + r * LK) = p;
      p.x = pkbf(acc1[r][0], acc1[r][1]);
      p.y = pkbf(acc1[r][2], acc1[r][3]);
      *(u32x2*)(xb + r * LK + 128) = p;
    }
  }
  wait_async0();                       // my share of W1 DMA complete
  __syncthreads();                     // X + all of W1 visible

  // ---- wave tiling for GEMMs: 2 (M) x 4 (N) waves, 4x4 WMMA tiles each ----
  const int mb = (wave & 1) * 64;
  const int nb = (wave >> 1) * 64;
  const int koff = hi * 8;

  v8f acc[16];
  const v8f vzero = {0.f, 0.f, 0.f, 0.f, 0.f, 0.f, 0.f, 0.f};

  // ---- GEMM1: H = relu(X @ W1 + b1) -> LDS bf16 ----
#pragma unroll
  for (int t = 0; t < 16; ++t) acc[t] = vzero;
  gemm_128x256(Xs + (mb + ln) * LK + koff, Ws + (nb + ln) * LK + koff, acc);

  {
    unsigned short* hb = Hs + (mb + 8 * hi) * LK + nb + ln;
#pragma unroll
    for (int ni = 0; ni < 4; ++ni) {
      const float bn = b1s[nb + ni * 16 + ln];
#pragma unroll
      for (int mi = 0; mi < 4; ++mi) {
#pragma unroll
        for (int r2 = 0; r2 < 4; ++r2) {
          float x0 = fmaxf(acc[mi * 4 + ni][2 * r2] + bn, 0.f);
          float x1 = fmaxf(acc[mi * 4 + ni][2 * r2 + 1] + bn, 0.f);
          const unsigned int pk = pkbf(x0, x1);
          hb[(mi * 16 + 2 * r2) * LK + ni * 16]     = (unsigned short)pk;
          hb[(mi * 16 + 2 * r2 + 1) * LK + ni * 16] = (unsigned short)(pk >> 16);
        }
      }
    }
  }
  __syncthreads();  // W1 reads done + H visible

  // ---- stage W2 over W1 (synchronous; clean codegen) ----
  {
    const unsigned short* src = wt2 + (size_t)j * DIM * DIM + wave * DIM + lane * 8;
    unsigned short* dst = Ws + wave * LK + lane * 8;
#pragma unroll
    for (int c = 0; c < 32; ++c)
      *(u32x4*)(dst + c * (8 * LK)) = *(const u32x4*)(src + c * (8 * DIM));
  }
  __syncthreads();

  // ---- GEMM2: out[..., 256:512] = relu(H @ W2 + b2) ----
#pragma unroll
  for (int t = 0; t < 16; ++t) acc[t] = vzero;
  gemm_128x256(Hs + (mb + ln) * LK + koff, Ws + (nb + ln) * LK + koff, acc);

  {
    float* ob = out + ((b0 + mb + 8 * hi) * PP + j) * 512 + 256 + nb + ln;
#pragma unroll
    for (int ni = 0; ni < 4; ++ni) {
      const float bn = b2s[nb + ni * 16 + ln];
#pragma unroll
      for (int mi = 0; mi < 4; ++mi) {
#pragma unroll
        for (int r = 0; r < 8; ++r) {
          float v = fmaxf(acc[mi * 4 + ni][r] + bn, 0.f);
          ob[(mi * 16 + r) * (PP * 512) + ni * 16] = v;
        }
      }
    }
  }
}

// ---------- host launcher ----------
extern "C" void kernel_launch(void* const* d_in, const int* in_sizes, int n_in,
                              void* d_out, int out_size, void* d_ws, size_t ws_size,
                              hipStream_t stream) {
  const float* inp   = (const float*)d_in[0];   // [8192,19,256]
  const float* aff   = (const float*)d_in[1];   // [8192,19,19]
  const float* w1    = (const float*)d_in[2];   // [19,256,256]
  const float* w2    = (const float*)d_in[3];   // [19,256,256]
  const float* b1    = (const float*)d_in[4];   // [19,256]
  const float* b2    = (const float*)d_in[5];   // [19,256]
  float* out         = (float*)d_out;           // [8192,19,512]

  unsigned short* wt1 = (unsigned short*)d_ws;                 // bf16 [19][n][k]
  unsigned short* wt2 = wt1 + (size_t)PP * DIM * DIM;

  (void)hipFuncSetAttribute((const void*)prop_fused,
                            hipFuncAttributeMaxDynamicSharedMemorySize, SMEM_BYTES);

  dim3 blk(256);
  prep_weights<<<dim3(8, 8, PP), blk, 0, stream>>>(w1, wt1);
  prep_weights<<<dim3(8, 8, PP), blk, 0, stream>>>(w2, wt2);
  prop_fused<<<dim3(PP, 8192 / BT), blk, SMEM_BYTES, stream>>>(
      inp, aff, b1, b2, wt1, wt2, out);
}
